// RSSM_14826227106218
// MI455X (gfx1250) — compile-verified, matrix operand order checked by reading, not run on hardware
//
#include <hip/hip_runtime.h>

// Problem constants (match reference)
#define B_DIM 256
#define T_DIM 64
#define A_DIM 64
#define E_DIM 1024
#define HID   1024
#define DET   2048
#define STO   256

typedef __attribute__((ext_vector_type(16))) __bf16       v16bf;
typedef __attribute__((ext_vector_type(8)))  float        v8f;
typedef __attribute__((ext_vector_type(4)))  unsigned int u32x4;

union Frag16 { v16bf v; u32x4 q[2]; };

__device__ __forceinline__ unsigned short f2bf(float x) {
  unsigned int u = __float_as_uint(x);
  unsigned int r = u + 0x7FFFu + ((u >> 16) & 1u);  // round-to-nearest-even
  return (unsigned short)(r >> 16);
}
__device__ __forceinline__ float sigm(float x) { return 1.f / (1.f + expf(-x)); }
__device__ __forceinline__ float softplus_(float x) { return (x > 20.f) ? x : log1pf(expf(x)); }

// CDNA5 async memory->LDS DMA (16B per lane), tracked with ASYNCcnt.
// vdst = 32-bit LDS byte offset (low 32 bits of a flat shared pointer),
// addr = 64-bit global address.
__device__ __forceinline__ void async_load_b128(unsigned lds_off, const void* gaddr) {
  asm volatile("global_load_async_to_lds_b128 %0, %1, off"
               :: "v"(lds_off), "v"((unsigned long long)(size_t)gaddr)
               : "memory");
}
__device__ __forceinline__ void wait_asynccnt0() {
  asm volatile("s_wait_asynccnt 0x0" ::: "memory");
}

// ---------------------------------------------------------------------------
// Tiled bf16 WMMA GEMM: C[M,N] = act( A[M,K] @ Bt[N,K]^T + bias )
//   A  : bf16 row-major, leading dim lda (elements)
//   Bt : bf16 row-major [N,K] (i.e. B transposed), leading dim ldb
//   Block tile: 64(M) x 128(N), K-tile 64. 256 threads = 8 waves, each wave
//   owns a 16x64 strip (4x v_wmma_f32_16x16x32_bf16 accumulators).
//   Tiles staged via global_load_async_to_lds_b128 (ASYNCcnt).
//   Requires: M%64==0, N%128==0, K%64==0 (true for all call sites here).
// ---------------------------------------------------------------------------
template <bool RELU>
__global__ void __launch_bounds__(256, 2)
k_gemm_bf16(const unsigned short* __restrict__ A, int lda,
            const unsigned short* __restrict__ Bt, int ldb,
            const float* __restrict__ bias,
            float* __restrict__ Cf, int ldcf,
            unsigned short* __restrict__ Cb, int ldcb,
            int K) {
  __shared__ unsigned short As[64 * 64];    // 8 KB
  __shared__ unsigned short Bs[128 * 64];   // 16 KB

  const int tid  = threadIdx.x;
  const int lane = tid & 31;
  const int wave = tid >> 5;       // 0..7
  const int wm   = wave & 3;       // M sub-tile 0..3
  const int wn   = wave >> 2;      // N sub-tile 0..1
  const int hl   = lane >> 4;      // lane half (0 or 1)
  const int l16  = lane & 15;

  const int m0 = blockIdx.x * 64;
  const int n0 = blockIdx.y * 128;

  const v8f vzero = {0.f, 0.f, 0.f, 0.f, 0.f, 0.f, 0.f, 0.f};
  v8f acc[4];
#pragma unroll
  for (int c = 0; c < 4; ++c) acc[c] = vzero;

  // Per-thread staging map (fixed across K-tiles): 2 chunks of A, 4 of B.
  int ar[2], ac[2], br[4], bc[4];
  unsigned aLds[2], bLds[4];
#pragma unroll
  for (int i = 0; i < 2; ++i) {
    int idx = (tid * 2 + i) * 8;           // 0..4095
    ar[i] = idx >> 6; ac[i] = idx & 63;
    aLds[i] = (unsigned)(size_t)&As[ar[i] * 64 + ac[i]];
  }
#pragma unroll
  for (int i = 0; i < 4; ++i) {
    int idx = (tid * 4 + i) * 8;           // 0..8191
    br[i] = idx >> 6; bc[i] = idx & 63;
    bLds[i] = (unsigned)(size_t)&Bs[br[i] * 64 + bc[i]];
  }

  for (int k0 = 0; k0 < K; k0 += 64) {
    // Async DMA stage: memory -> LDS, 16B per lane per op (ASYNCcnt tracked)
#pragma unroll
    for (int i = 0; i < 2; ++i)
      async_load_b128(aLds[i], &A[(size_t)(m0 + ar[i]) * lda + k0 + ac[i]]);
#pragma unroll
    for (int i = 0; i < 4; ++i)
      async_load_b128(bLds[i], &Bt[(size_t)(n0 + br[i]) * ldb + k0 + bc[i]]);
    // Prefetch next K-tile of the (L2-resident) weight stream
    if (k0 + 64 < K) {
      __builtin_prefetch(&Bt[(size_t)(n0 + (tid >> 1)) * ldb + k0 + 64 + (tid & 1) * 32], 0, 1);
      __builtin_prefetch(&A[(size_t)(m0 + (tid & 63)) * lda + k0 + 64], 0, 1);
    }
    wait_asynccnt0();
    __syncthreads();

#pragma unroll
    for (int kk = 0; kk < 64; kk += 32) {
      // A fragment: row = wm*16 + l16; K pattern per ISA 7.12.2 (16-bit A 16x32)
      Frag16 af;
      const unsigned short* arow = &As[(wm * 16 + l16) * 64 + kk + hl * 8];
      af.q[0] = *(const u32x4*)(arow);
      af.q[1] = *(const u32x4*)(arow + 16);
      // Preload all 4 B fragments, then issue the WMMA chain back-to-back so
      // the matrix pipe overlaps with outstanding LDS loads (partial dscnt).
      Frag16 bfr[4];
#pragma unroll
      for (int c = 0; c < 4; ++c) {
        const unsigned short* brow = &Bs[(wn * 64 + c * 16 + l16) * 64 + kk + hl * 8];
        bfr[c].q[0] = *(const u32x4*)(brow);
        bfr[c].q[1] = *(const u32x4*)(brow + 16);
      }
#pragma unroll
      for (int c = 0; c < 4; ++c) {
        acc[c] = __builtin_amdgcn_wmma_f32_16x16x32_bf16(
            false, af.v, false, bfr[c].v, (short)0, acc[c], false, false);
      }
    }
    __syncthreads();
  }

  // Epilogue: C/D layout — lane l16 = N, VGPR j = M (j + 8*half)
#pragma unroll
  for (int c = 0; c < 4; ++c) {
    const int col = n0 + wn * 64 + c * 16 + l16;
    const float bv = bias ? bias[col] : 0.f;
#pragma unroll
    for (int j = 0; j < 8; ++j) {
      const int row = m0 + wm * 16 + hl * 8 + j;
      float v = acc[c][j] + bv;
      if (RELU) v = v > 0.f ? v : 0.f;
      if (Cf) Cf[(size_t)row * ldcf + col] = v;
      if (Cb) Cb[(size_t)row * ldcb + col] = f2bf(v);
    }
  }
}

// ---------------------------------------------------------------------------
// Setup / elementwise kernels
// ---------------------------------------------------------------------------
__global__ void k_zero_b32(unsigned int* p, int n) {
  int i = blockIdx.x * blockDim.x + threadIdx.x;
  if (i < n) p[i] = 0u;
}
// Wt[i] = bf16(W[i])  (W already [N,K])
__global__ void k_conv_direct(const float* __restrict__ W, unsigned short* __restrict__ Wt, int n) {
  int i = blockIdx.x * blockDim.x + threadIdx.x;
  if (i < n) Wt[i] = f2bf(W[i]);
}
// Wt[n*K + k] = bf16(W[k*N + n])  (W is [K,N] row-major)
__global__ void k_conv_transpose(const float* __restrict__ W, unsigned short* __restrict__ Wt,
                                 int K, int N) {
  int i = blockIdx.x * blockDim.x + threadIdx.x;
  if (i < K * N) {
    int k = i / N, n = i - k * N;
    Wt[(size_t)n * K + k] = f2bf(W[i]);
  }
}
// rnncat[b*320 + 256 + a] = bf16(t==0 ? 0 : actions[b, t-1, a])
__global__ void k_set_actions(const float* __restrict__ actions, unsigned short* __restrict__ rnncat, int t) {
  int i = blockIdx.x * blockDim.x + threadIdx.x;       // B*A
  int b = i >> 6, a = i & 63;
  float v = (t == 0) ? 0.f : actions[((size_t)b * T_DIM + (t - 1)) * A_DIM + a];
  rnncat[(size_t)b * (STO + A_DIM) + STO + a] = f2bf(v);
}
// postcat[b*3072 + 2048 + e] = bf16(obs[b, t, e])
__global__ void k_set_obs(const float* __restrict__ obs, unsigned short* __restrict__ postcat, int t) {
  int i = blockIdx.x * blockDim.x + threadIdx.x;       // B*E
  int b = i >> 10, e = i & 1023;
  postcat[(size_t)b * (DET + E_DIM) + DET + e] =
      f2bf(obs[((size_t)b * T_DIM + t) * E_DIM + e]);
}
// GRU gate combine; writes new h to hstate (fp32), postcat h-part (bf16), outH[b,t,:]
__global__ void k_gru(const float* __restrict__ gi, const float* __restrict__ gh,
                      float* __restrict__ hstate, unsigned short* __restrict__ postcat,
                      float* __restrict__ outH, int t) {
  int i = blockIdx.x * blockDim.x + threadIdx.x;       // B*DET
  int b = i >> 11, n = i & 2047;
  size_t gb = (size_t)b * 3 * DET;
  float r = sigm(gi[gb + n] + gh[gb + n]);
  float u = sigm(gi[gb + DET + n] + gh[gb + DET + n]);
  float nn = tanhf(gi[gb + 2 * DET + n] + r * gh[gb + 2 * DET + n]);
  float h = (1.f - u) * nn + u * hstate[(size_t)b * DET + n];
  hstate[(size_t)b * DET + n] = h;
  postcat[(size_t)b * (DET + E_DIM) + n] = f2bf(h);
  outH[((size_t)b * T_DIM + t) * DET + n] = h;
}
// posterior: rsample + outputs; stash bf16(z) into rnncat for next step
__global__ void k_posterior(const float* __restrict__ pms, const float* __restrict__ noise,
                            float* __restrict__ outZ, float* __restrict__ outPM,
                            float* __restrict__ outPS, unsigned short* __restrict__ rnncat, int t) {
  int i = blockIdx.x * blockDim.x + threadIdx.x;       // B*STO
  int b = i >> 8, n = i & 255;
  float pm = pms[(size_t)b * 2 * STO + n];
  float ps = softplus_(pms[(size_t)b * 2 * STO + STO + n]) + 0.1f;
  float z = pm + ps * noise[((size_t)b * T_DIM + t) * STO + n];
  size_t o = ((size_t)b * T_DIM + t) * STO + n;
  outZ[o] = z; outPM[o] = pm; outPS[o] = ps;
  rnncat[(size_t)b * (STO + A_DIM) + n] = f2bf(z);
}
__global__ void k_prior(const float* __restrict__ qms, float* __restrict__ outQM,
                        float* __restrict__ outQS, int t) {
  int i = blockIdx.x * blockDim.x + threadIdx.x;       // B*STO
  int b = i >> 8, n = i & 255;
  size_t o = ((size_t)b * T_DIM + t) * STO + n;
  outQM[o] = qms[(size_t)b * 2 * STO + n];
  outQS[o] = softplus_(qms[(size_t)b * 2 * STO + STO + n]) + 0.1f;
}

// ---------------------------------------------------------------------------
extern "C" void kernel_launch(void* const* d_in, const int* in_sizes, int n_in,
                              void* d_out, int out_size, void* d_ws, size_t ws_size,
                              hipStream_t stream) {
  (void)in_sizes; (void)n_in; (void)out_size; (void)ws_size;
  const float* obs        = (const float*)d_in[0];
  const float* actions    = (const float*)d_in[1];
  const float* noise      = (const float*)d_in[4];
  const float* W_rnn_in   = (const float*)d_in[5];
  const float* b_rnn_in   = (const float*)d_in[6];
  const float* gru_Wih    = (const float*)d_in[7];
  const float* gru_Whh    = (const float*)d_in[8];
  const float* gru_bih    = (const float*)d_in[9];
  const float* gru_bhh    = (const float*)d_in[10];
  const float* W_post_in  = (const float*)d_in[11];
  const float* b_post_in  = (const float*)d_in[12];
  const float* W_post_ms  = (const float*)d_in[13];
  const float* b_post_ms  = (const float*)d_in[14];
  const float* W_prior_in = (const float*)d_in[15];
  const float* b_prior_in = (const float*)d_in[16];
  const float* W_prior_ms = (const float*)d_in[17];
  const float* b_prior_ms = (const float*)d_in[18];

  // Output tuple layout (flat, in return order)
  float* outH  = (float*)d_out;
  float* outZ  = outH  + (size_t)B_DIM * T_DIM * DET;
  float* outPM = outZ  + (size_t)B_DIM * T_DIM * STO;
  float* outPS = outPM + (size_t)B_DIM * T_DIM * STO;
  float* outQM = outPS + (size_t)B_DIM * T_DIM * STO;
  float* outQS = outQM + (size_t)B_DIM * T_DIM * STO;

  // Workspace carve-out (256B aligned)
  char* ws = (char*)d_ws;
  size_t off = 0;
  auto alloc = [&](size_t bytes) -> char* {
    char* p = ws + off;
    off = (off + bytes + 255) & ~(size_t)255;
    return p;
  };
  // bf16 weights (51 MB total — resident in the 192 MB L2 across all 64 steps)
  unsigned short* Wrnn_t   = (unsigned short*)alloc((size_t)HID * (STO + A_DIM) * 2);
  unsigned short* Wih_b    = (unsigned short*)alloc((size_t)3 * DET * HID * 2);
  unsigned short* Whh_b    = (unsigned short*)alloc((size_t)3 * DET * DET * 2);
  unsigned short* Wpin_t   = (unsigned short*)alloc((size_t)HID * (DET + E_DIM) * 2);
  unsigned short* Wpms_t   = (unsigned short*)alloc((size_t)2 * STO * HID * 2);
  unsigned short* Wqin_t   = (unsigned short*)alloc((size_t)HID * DET * 2);
  unsigned short* Wqms_t   = (unsigned short*)alloc((size_t)2 * STO * HID * 2);
  // activations
  unsigned short* rnncat   = (unsigned short*)alloc((size_t)B_DIM * (STO + A_DIM) * 2);
  unsigned short* rnnhid   = (unsigned short*)alloc((size_t)B_DIM * HID * 2);
  float*          gi       = (float*)alloc((size_t)B_DIM * 3 * DET * 4);
  float*          gh       = (float*)alloc((size_t)B_DIM * 3 * DET * 4);
  float*          hstate   = (float*)alloc((size_t)B_DIM * DET * 4);
  unsigned short* postcat  = (unsigned short*)alloc((size_t)B_DIM * (DET + E_DIM) * 2);
  unsigned short* pf       = (unsigned short*)alloc((size_t)B_DIM * HID * 2);
  float*          pms      = (float*)alloc((size_t)B_DIM * 2 * STO * 4);
  unsigned short* rf       = (unsigned short*)alloc((size_t)B_DIM * HID * 2);
  float*          qms      = (float*)alloc((size_t)B_DIM * 2 * STO * 4);

  const int TPB = 256;
  auto blocks = [](size_t n) { return dim3((unsigned)((n + 255) / 256)); };

  // --- One-time weight conversion to bf16 [N][K] layout ---
  k_conv_transpose<<<blocks((size_t)(STO + A_DIM) * HID), TPB, 0, stream>>>(W_rnn_in, Wrnn_t, STO + A_DIM, HID);
  k_conv_direct  <<<blocks((size_t)3 * DET * HID), TPB, 0, stream>>>(gru_Wih, Wih_b, 3 * DET * HID);
  k_conv_direct  <<<blocks((size_t)3 * DET * DET), TPB, 0, stream>>>(gru_Whh, Whh_b, 3 * DET * DET);
  k_conv_transpose<<<blocks((size_t)(DET + E_DIM) * HID), TPB, 0, stream>>>(W_post_in, Wpin_t, DET + E_DIM, HID);
  k_conv_transpose<<<blocks((size_t)HID * 2 * STO), TPB, 0, stream>>>(W_post_ms, Wpms_t, HID, 2 * STO);
  k_conv_transpose<<<blocks((size_t)DET * HID), TPB, 0, stream>>>(W_prior_in, Wqin_t, DET, HID);
  k_conv_transpose<<<blocks((size_t)HID * 2 * STO), TPB, 0, stream>>>(W_prior_ms, Wqms_t, HID, 2 * STO);

  // --- Zero initial state (h0 = 0, z0 = 0; harness poisons ws) ---
  k_zero_b32<<<blocks((size_t)B_DIM * (STO + A_DIM) / 2), TPB, 0, stream>>>((unsigned int*)rnncat, B_DIM * (STO + A_DIM) / 2);
  k_zero_b32<<<blocks((size_t)B_DIM * (DET + E_DIM) / 2), TPB, 0, stream>>>((unsigned int*)postcat, B_DIM * (DET + E_DIM) / 2);
  k_zero_b32<<<blocks((size_t)B_DIM * DET), TPB, 0, stream>>>((unsigned int*)hstate, B_DIM * DET);

  const dim3 gN1024(B_DIM / 64, HID / 128);      // (4, 8)
  const dim3 gN6144(B_DIM / 64, 3 * DET / 128);  // (4, 48)
  const dim3 gN512 (B_DIM / 64, 2 * STO / 128);  // (4, 4)

  for (int t = 0; t < T_DIM; ++t) {
    // [z_{t-1} ; a_{t-1}] -> bf16 (z part written by previous k_posterior)
    k_set_actions<<<blocks((size_t)B_DIM * A_DIM), TPB, 0, stream>>>(actions, rnncat, t);
    // rnn_in = relu([z;a] @ W_rnn_in + b)  -> bf16
    k_gemm_bf16<true><<<gN1024, TPB, 0, stream>>>(rnncat, STO + A_DIM, Wrnn_t, STO + A_DIM,
                                                  b_rnn_in, nullptr, 0, rnnhid, HID, STO + A_DIM);
    // gi = rnn_in @ Wih^T + bih ; gh = h_prev @ Whh^T + bhh   (h_prev lives in postcat, lda=3072)
    k_gemm_bf16<false><<<gN6144, TPB, 0, stream>>>(rnnhid, HID, Wih_b, HID,
                                                   gru_bih, gi, 3 * DET, nullptr, 0, HID);
    k_gemm_bf16<false><<<gN6144, TPB, 0, stream>>>(postcat, DET + E_DIM, Whh_b, DET,
                                                   gru_bhh, gh, 3 * DET, nullptr, 0, DET);
    // GRU combine -> h_t (fp32 state, bf16 into postcat, fp32 into outH)
    k_gru<<<blocks((size_t)B_DIM * DET), TPB, 0, stream>>>(gi, gh, hstate, postcat, outH, t);
    // obs_t -> bf16 tail of postcat
    k_set_obs<<<blocks((size_t)B_DIM * E_DIM), TPB, 0, stream>>>(obs, postcat, t);
    // posterior branch
    k_gemm_bf16<true><<<gN1024, TPB, 0, stream>>>(postcat, DET + E_DIM, Wpin_t, DET + E_DIM,
                                                  b_post_in, nullptr, 0, pf, HID, DET + E_DIM);
    k_gemm_bf16<false><<<gN512, TPB, 0, stream>>>(pf, HID, Wpms_t, HID,
                                                  b_post_ms, pms, 2 * STO, nullptr, 0, HID);
    k_posterior<<<blocks((size_t)B_DIM * STO), TPB, 0, stream>>>(pms, noise, outZ, outPM, outPS, rnncat, t);
    // prior branch (uses h_t part of postcat, K=2048)
    k_gemm_bf16<true><<<gN1024, TPB, 0, stream>>>(postcat, DET + E_DIM, Wqin_t, DET,
                                                  b_prior_in, nullptr, 0, rf, HID, DET);
    k_gemm_bf16<false><<<gN512, TPB, 0, stream>>>(rf, HID, Wqms_t, HID,
                                                  b_prior_ms, qms, 2 * STO, nullptr, 0, HID);
    k_prior<<<blocks((size_t)B_DIM * STO), TPB, 0, stream>>>(qms, outQM, outQS, t);
  }
}
